// Diffusion_3418793968193
// MI455X (gfx1250) — matrix operand order repaired and checked by reading, not compiled, not run on hardware
//
#include <hip/hip_runtime.h>
#include <cstddef>

// Problem constants
#define NB 8
#define NN 2048
#define TCC 96
#define TFF 64
#define HGG 64
#define NL 4

typedef __attribute__((ext_vector_type(16))) __bf16 bf16x16;
typedef __attribute__((ext_vector_type(8)))  __bf16 bf16x8;
typedef __attribute__((ext_vector_type(8)))  float  f32x8;

#define LDSS 40  // padded LDS row stride (bf16 elems): 80B, 16B-aligned, odd bank phase

__device__ __forceinline__ __bf16 f2bf(float f) {
  union { float f; unsigned u; } c; c.f = f;
  unsigned r = (c.u + 0x7FFFu + ((c.u >> 16) & 1u)) >> 16;  // RNE truncate
  union { unsigned short s; __bf16 b; } o; o.s = (unsigned short)r;
  return o.b;
}

// ---- WMMA fragment loaders (wave32 layouts per CDNA5 ISA 7.12.2) ----
// A 16x32 bf16: lane L -> row M=L&15; K elems: {kb..kb+7, kb+16..kb+23}, kb=8*(L>=16)
__device__ __forceinline__ bf16x16 load_a_frag(const __bf16* tile, int lane) {
  int row = lane & 15;
  int kb  = (lane >> 4) << 3;
  const __bf16* p = tile + row * LDSS + kb;
  bf16x8 lo = *(const bf16x8*)(p);
  bf16x8 hi = *(const bf16x8*)(p + 16);
  bf16x16 r;
#pragma unroll
  for (int i = 0; i < 8; ++i) { r[i] = lo[i]; r[i + 8] = hi[i]; }
  return r;
}
// B 32x16 bf16 from an LDS tile stored transposed bT[N][K]:
// lane L -> col N=L&15; K elems: {kb..kb+7, kb+16..kb+23}, kb=8*(L>=16)
__device__ __forceinline__ bf16x16 load_b_frag(const __bf16* tileT, int lane) {
  int col = lane & 15;
  int kb  = (lane >> 4) << 3;
  const __bf16* p = tileT + col * LDSS + kb;
  bf16x8 lo = *(const bf16x8*)(p);
  bf16x8 hi = *(const bf16x8*)(p + 16);
  bf16x16 r;
#pragma unroll
  for (int i = 0; i < 8; ++i) { r[i] = lo[i]; r[i + 8] = hi[i]; }
  return r;
}

// =========================== generic bf16 WMMA GEMM ===========================
// D[m, col] = sum_k A[m,k]*B[k,col] (+ mode epilogue). Cols enumerate (b,t) or (s,b).
// Workgroup tile 128x128 (8 waves as 4Mx2N, each wave 32x64 = 8 WMMAs / K-step).
enum { MODE_CONV = 0, MODE_PROJ = 1, MODE_OUT = 2, MODE_GI = 3 };

struct GemmArgs {
  const float* A;     // [M x K] row-major weights (fp32)
  const float* bias;  // per-row bias
  const float* Bsrc;  // activations
  const float* res;   // PROJ: residual y_in; OUT: noise
  float*       out;   // output tensor, or loss scalar for OUT
  int M, K, dil;
  float scale;        // OUT: 1/count
};

template <int MODE>
__global__ __launch_bounds__(256) void wmma_gemm(GemmArgs g) {
  __shared__ __align__(16) __bf16 a_lds[128 * LDSS];
  __shared__ __align__(16) __bf16 b_lds[128 * LDSS];
  const int tid  = threadIdx.x;
  const int lane = tid & 31, wid = tid >> 5;
  const int wm = wid >> 1, wn = wid & 1;   // 4 x 2 wave grid, each wave 32(M) x 64(N)
  const int row0 = blockIdx.x * 128;
  const int col0 = blockIdx.y * 128;

  f32x8 acc[2][4];
#pragma unroll
  for (int i = 0; i < 2; ++i)
#pragma unroll
    for (int j = 0; j < 4; ++j)
#pragma unroll
      for (int q = 0; q < 8; ++q) acc[i][j][q] = 0.f;

  for (int k0 = 0; k0 < g.K; k0 += 32) {
    __syncthreads();
    // ---- stage A tile 128x32 fp32 -> bf16 (coalesced rows) ----
    for (int e = tid; e < 128 * 32; e += 256) {
      int r = e >> 5, kk = e & 31;
      int gr = row0 + r;
      float v = 0.f;
      if (MODE != MODE_GI || gr < g.M) {
        const float* ap = g.A + (size_t)gr * g.K + (k0 + kk);
        v = *ap;
        __builtin_prefetch(ap + 32, 0, 1);  // next K-chunk -> global_prefetch_b8
      }
      a_lds[r * LDSS + kk] = f2bf(v);
    }
    // ---- stage B tile transposed bT[128 n][32 k] with mode-specific gather ----
    for (int e = tid; e < 128 * 32; e += 256) {
      int n = e >> 5, kk = e & 31;
      int kidx = k0 + kk;
      int col  = col0 + n;
      float v = 0.f;
      if (MODE == MODE_CONV) {
        int c = kidx / 3, k3 = kidx - c * 3;
        int b = col >> 6, t = col & 63;
        int t2 = t - (2 - k3) * g.dil;  // causal dilated tap
        if (t2 >= 0) v = g.Bsrc[((size_t)(b << 11) + c) * 64 + t2];
      } else if (MODE == MODE_GI) {
        int s = col >> 3, b = col & 7;  // col = s*8 + b
        v = g.Bsrc[((size_t)(b << 11) + kidx) * 96 + s];  // ctx[b,n,s]
      } else {
        int b = col >> 6, t = col & 63;
        v = g.Bsrc[((size_t)(b << 11) + kidx) * 64 + t];
      }
      b_lds[n * LDSS + kk] = f2bf(v);
    }
    __syncthreads();
    bf16x16 af[2], bq[4];
#pragma unroll
    for (int i = 0; i < 2; ++i)
      af[i] = load_a_frag(&a_lds[(wm * 32 + i * 16) * LDSS], lane);
#pragma unroll
    for (int j = 0; j < 4; ++j)
      bq[j] = load_b_frag(&b_lds[(wn * 64 + j * 16) * LDSS], lane);
#pragma unroll
    for (int i = 0; i < 2; ++i)
#pragma unroll
      for (int j = 0; j < 4; ++j)
        acc[i][j] = __builtin_amdgcn_wmma_f32_16x16x32_bf16(
            false, af[i], false, bq[j], (short)0, acc[i][j], false, false);
  }

  // ---- epilogue: D layout lane->col N=L&15, elem r -> row r + 8*(L>=16) ----
  const int nl = lane & 15;
  const int mo = (lane >> 4) << 3;
  float lsum = 0.f;
#pragma unroll
  for (int i = 0; i < 2; ++i)
#pragma unroll
    for (int j = 0; j < 4; ++j)
#pragma unroll
      for (int r = 0; r < 8; ++r) {
        int m   = row0 + wm * 32 + i * 16 + r + mo;
        int col = col0 + wn * 64 + j * 16 + nl;
        float v = acc[i][j][r];
        if (MODE == MODE_GI) {
          if (m < g.M) g.out[(size_t)col * 192 + m] = v + g.bias[m];
        } else {
          int b = col >> 6, t = col & 63;
          size_t o = ((size_t)(b << 11) + m) * 64 + t;
          if (MODE == MODE_CONV) {
            g.out[o] = fmaxf(v + g.bias[m], 0.f);              // ReLU
          } else if (MODE == MODE_PROJ) {
            g.out[o] = v + g.bias[m] + g.res[o];               // residual add
          } else {                                             // MODE_OUT: MSE
            float d = v + g.bias[m] - g.res[o];
            lsum += d * d;
          }
        }
      }
  if (MODE == MODE_OUT) {
#pragma unroll
    for (int off = 16; off; off >>= 1) lsum += __shfl_xor(lsum, off, 32);
    if (lane == 0) atomicAdd(g.out, lsum * g.scale);
  }
}

// =========================== fused flash-GAT (WMMA) ===========================
// Y[b,i,t] = softmax_j(lrelu(ei[i]+ej[j])) @ H[b,j,t] + cond[b,i]
// Row max = lrelu(ei + max_j ej) since lrelu is monotonic -> single pass.
__global__ __launch_bounds__(256) void gat_kernel(const float* __restrict__ H,
                                                  const float* __restrict__ ei,
                                                  const float* __restrict__ ej,
                                                  const float* __restrict__ mxej,
                                                  const float* __restrict__ cond,
                                                  float* __restrict__ Y) {
  __shared__ __align__(16) __bf16 a_lds[128 * LDSS];
  __shared__ __align__(16) __bf16 b_lds[64 * LDSS];
  __shared__ float zsum[128];
  const int tid = threadIdx.x, lane = tid & 31, wid = tid >> 5;
  const int wm = wid >> 1, wn = wid & 1;   // wave 32(i) x 32(t)
  const int i0 = blockIdx.x * 128;
  const int b  = blockIdx.y;
  const int rowA = tid & 127;
  const int half = tid >> 7;
  if (tid < 128) zsum[tid] = 0.f;
  const float eii = ei[b * 2048 + i0 + rowA];
  float mm = eii + mxej[b];
  mm = (mm > 0.f) ? mm : 0.2f * mm;
  float zpart = 0.f;

  f32x8 acc[2][2];
#pragma unroll
  for (int i = 0; i < 2; ++i)
#pragma unroll
    for (int j = 0; j < 2; ++j)
#pragma unroll
      for (int q = 0; q < 8; ++q) acc[i][j][q] = 0.f;

  for (int j0 = 0; j0 < 2048; j0 += 32) {
    __syncthreads();
    // generate A tile (attention weights) directly in bf16
#pragma unroll
    for (int q = 0; q < 16; ++q) {
      int jj = half * 16 + q;
      float e = eii + ej[b * 2048 + j0 + jj];
      e = (e > 0.f) ? e : 0.2f * e;      // leaky_relu(0.2)
      float w = __expf(e - mm);
      a_lds[rowA * LDSS + jj] = f2bf(w);
      zpart += w;
    }
    // stage B = H[b, j0:j0+32, :] transposed to bT[t][jj]
    for (int e2 = tid; e2 < 64 * 32; e2 += 256) {
      int t = e2 >> 5, jj = e2 & 31;
      b_lds[t * LDSS + jj] = f2bf(H[((size_t)(b << 11) + (j0 + jj)) * 64 + t]);
    }
    __syncthreads();
    bf16x16 af0 = load_a_frag(&a_lds[(wm * 32) * LDSS], lane);
    bf16x16 af1 = load_a_frag(&a_lds[(wm * 32 + 16) * LDSS], lane);
    bf16x16 bq0 = load_b_frag(&b_lds[(wn * 32) * LDSS], lane);
    bf16x16 bq1 = load_b_frag(&b_lds[(wn * 32 + 16) * LDSS], lane);
    acc[0][0] = __builtin_amdgcn_wmma_f32_16x16x32_bf16(false, af0, false, bq0, (short)0, acc[0][0], false, false);
    acc[0][1] = __builtin_amdgcn_wmma_f32_16x16x32_bf16(false, af0, false, bq1, (short)0, acc[0][1], false, false);
    acc[1][0] = __builtin_amdgcn_wmma_f32_16x16x32_bf16(false, af1, false, bq0, (short)0, acc[1][0], false, false);
    acc[1][1] = __builtin_amdgcn_wmma_f32_16x16x32_bf16(false, af1, false, bq1, (short)0, acc[1][1], false, false);
  }
  atomicAdd(&zsum[rowA], zpart);  // ds_add_f32
  __syncthreads();

  const int nl = lane & 15;
  const int mo = (lane >> 4) << 3;
#pragma unroll
  for (int i = 0; i < 2; ++i) {
    // one fast reciprocal per row-fragment pair (softmax scale), v_rcp_f32
    float rz0 = __builtin_amdgcn_rcpf(zsum[wm * 32 + i * 16 + 0 + mo]);
#pragma unroll
    for (int j = 0; j < 2; ++j)
#pragma unroll
      for (int r = 0; r < 8; ++r) {
        int ml = wm * 32 + i * 16 + r + mo;
        float rz = (r == 0) ? rz0 : __builtin_amdgcn_rcpf(zsum[ml]);
        int t  = wn * 32 + j * 16 + nl;
        int ig = i0 + ml;
        Y[((size_t)(b << 11) + ig) * 64 + t] = acc[i][j][r] * rz + cond[b * 2048 + ig];
      }
  }
}

// =========================== small VALU kernels ===========================
__global__ void zero_loss_kernel(float* p) { if (threadIdx.x == 0) p[0] = 0.f; }

// cosine diffusion schedule: alphas_bar on device (per-batch)
__global__ void ab_kernel(const int* __restrict__ k, float* sab, float* s1ab) {
  int b = threadIdx.x;
  if (b >= NB) return;
  const float T = 100.f, s = 0.008f, PI = 3.14159265358979323846f;
  float f0 = cosf((s / (1.f + s)) * PI * 0.5f); f0 = f0 * f0;
  float prev = 1.f, cum = 1.f;
  int kk = k[b];
  for (int i = 1; i <= kk + 1; ++i) {
    float fi = cosf((((float)i / T + s) / (1.f + s)) * PI * 0.5f); fi = fi * fi;
    float ab = fi / f0;
    float beta = 1.f - ab / prev;
    beta = fminf(fmaxf(beta, 1e-6f), 0.999f);
    cum *= (1.f - beta);
    prev = ab;
  }
  sab[b] = sqrtf(cum);
  s1ab[b] = sqrtf(1.f - cum);
}

__global__ __launch_bounds__(256) void qsample_kernel(const float* __restrict__ fut,
                                                      const float* __restrict__ noise,
                                                      const float* __restrict__ sab,
                                                      const float* __restrict__ s1ab,
                                                      float* __restrict__ xk) {
  int idx = blockIdx.x * 256 + threadIdx.x;   // 8*2048*64 = 1048576 elems exactly
  int b = idx >> 17;
  xk[idx] = sab[b] * fut[idx] + s1ab[b] * noise[idx];
}

// GRU recurrence: 96 sequential steps, gi precomputed by WMMA GEMM
__global__ __launch_bounds__(512) void gru_scan(const float* __restrict__ gi,
                                                const float* __restrict__ whh,
                                                const float* __restrict__ bhh,
                                                float* __restrict__ ht_out) {
  __shared__ float h[NB * HGG];        // 512
  __shared__ float gh[NB * 192];       // 1536
  __shared__ float W[192 * HGG];       // 48KB
  __shared__ float bh[192];
  int tid = threadIdx.x;
  for (int e = tid; e < 192 * 64; e += 512) W[e] = whh[e];
  if (tid < 192) bh[tid] = bhh[tid];
  h[tid] = 0.f;
  __syncthreads();
  for (int st = 0; st < TCC; ++st) {
    for (int o = tid; o < 1536; o += 512) {
      int b = o / 192, g = o - b * 192;
      const float* wr = &W[g * 64];
      const float* hb = &h[b * 64];
      float a = bh[g];
#pragma unroll
      for (int j = 0; j < 64; ++j) a += hb[j] * wr[j];
      gh[o] = a;
    }
    __syncthreads();
    {
      int b = tid >> 6, j = tid & 63;
      const float* gis = &gi[(size_t)(st * 8 + b) * 192];
      float ir = gis[j], iz = gis[64 + j], in = gis[128 + j];
      float hr = gh[b * 192 + j], hz = gh[b * 192 + 64 + j], hn = gh[b * 192 + 128 + j];
      float r = 1.f / (1.f + __expf(-(ir + hr)));
      float z = 1.f / (1.f + __expf(-(iz + hz)));
      float n = tanhf(in + r * hn);
      float hv = (1.f - z) * n + z * h[tid];
      __syncthreads();
      h[tid] = hv;
    }
    __syncthreads();
  }
  ht_out[tid] = h[tid];
}

__global__ __launch_bounds__(256) void cond_kernel(const float* __restrict__ ht,
                                                   const float* __restrict__ htp_w,
                                                   const float* __restrict__ htp_b,
                                                   float* __restrict__ cond) {
  __shared__ float hs[NB * HGG];
  for (int e = threadIdx.x; e < 512; e += 256) hs[e] = ht[e];
  __syncthreads();
  int idx = blockIdx.x * 256 + threadIdx.x;  // 16384
  int b = idx >> 11, n = idx & 2047;
  const float* hb = &hs[b * 64];
  const float* w  = &htp_w[(size_t)n * 64];
  float a = htp_b[n];
#pragma unroll
  for (int j = 0; j < 64; ++j) a += hb[j] * w[j];
  cond[idx] = a;
}

// H = y @ gat_w^T per (b,n); also ei = H.a_i, ej = H.a_j
__global__ __launch_bounds__(256) void hei_kernel(const float* __restrict__ y,
                                                  const float* __restrict__ gat_w,
                                                  const float* __restrict__ gat_a,
                                                  float* __restrict__ H,
                                                  float* __restrict__ ei,
                                                  float* __restrict__ ej) {
  __shared__ float gw[64 * 64];
  __shared__ float ga[128];
  for (int e = threadIdx.x; e < 4096; e += 256) gw[e] = gat_w[e];
  if (threadIdx.x < 128) ga[threadIdx.x] = gat_a[threadIdx.x];
  __syncthreads();
  int idx = blockIdx.x * 256 + threadIdx.x;  // b*2048+n
  float yr[64];
  const float* yp = y + (size_t)idx * 64;
#pragma unroll
  for (int t = 0; t < 64; ++t) yr[t] = yp[t];
  float se = 0.f, sj = 0.f;
  float* Hp = H + (size_t)idx * 64;
  for (int o = 0; o < 64; ++o) {
    const float* w = &gw[o * 64];
    float hv = 0.f;
#pragma unroll
    for (int t = 0; t < 64; ++t) hv += yr[t] * w[t];
    Hp[o] = hv;
    se += hv * ga[o];
    sj += hv * ga[64 + o];
  }
  ei[idx] = se;
  ej[idx] = sj;
}

__global__ __launch_bounds__(256) void maxej_kernel(const float* __restrict__ ej, float* mx) {
  __shared__ float red[256];
  int b = blockIdx.x;
  float m = -1e30f;
  for (int j = threadIdx.x; j < 2048; j += 256) m = fmaxf(m, ej[b * 2048 + j]);
  red[threadIdx.x] = m;
  __syncthreads();
  for (int s2 = 128; s2; s2 >>= 1) {
    if (threadIdx.x < s2) red[threadIdx.x] = fmaxf(red[threadIdx.x], red[threadIdx.x + s2]);
    __syncthreads();
  }
  if (threadIdx.x == 0) mx[b] = red[0];
}

// =========================== launch ===========================
extern "C" void kernel_launch(void* const* d_in, const int* in_sizes, int n_in,
                              void* d_out, int out_size, void* d_ws, size_t ws_size,
                              hipStream_t stream) {
  (void)in_sizes; (void)n_in; (void)out_size; (void)ws_size;
  const float* ctx    = (const float*)d_in[0];
  const float* fut    = (const float*)d_in[1];
  const float* noise  = (const float*)d_in[2];
  const float* conv_w = (const float*)d_in[3];
  const float* conv_b = (const float*)d_in[4];
  const float* proj_w = (const float*)d_in[5];
  const float* proj_b = (const float*)d_in[6];
  const float* gat_w  = (const float*)d_in[7];
  const float* gat_a  = (const float*)d_in[8];
  const float* out_w  = (const float*)d_in[9];
  const float* out_b  = (const float*)d_in[10];
  const float* htp_w  = (const float*)d_in[11];
  const float* htp_b  = (const float*)d_in[12];
  const float* wih    = (const float*)d_in[13];
  const float* whh    = (const float*)d_in[14];
  const float* bih    = (const float*)d_in[15];
  const float* bhh    = (const float*)d_in[16];
  const int*   kst    = (const int*)d_in[17];
  float* loss = (float*)d_out;

  // workspace carve (floats): ~21.5 MB total
  float* ws   = (float*)d_ws;
  float* sab  = ws;                   // 8
  float* s1ab = ws + 8;               // 8
  float* gi   = ws + 16;              // 768*192
  float* ht   = gi + 768 * 192;       // 512
  float* cond = ht + 512;             // 16384
  float* y0   = cond + 16384;         // 1048576
  float* y1   = y0 + 1048576;         // 1048576
  float* hbuf = y1 + 1048576;         // 1048576
  float* Hm   = hbuf + 1048576;       // 1048576
  float* ei   = Hm + 1048576;         // 16384
  float* ej   = ei + 16384;           // 16384
  float* mxej = ej + 16384;           // 8
  float* Yb   = mxej + 8;             // 1048576

  zero_loss_kernel<<<1, 32, 0, stream>>>(loss);
  ab_kernel<<<1, 32, 0, stream>>>(kst, sab, s1ab);

  // GRU input GEMM: gi[s,b,:] = ctx[:,:,s] @ wih^T + bih   (M=192, K=2048, cols=768)
  {
    GemmArgs g{wih, bih, ctx, nullptr, gi, 192, 2048, 0, 0.f};
    wmma_gemm<MODE_GI><<<dim3(2, 6), 256, 0, stream>>>(g);
  }
  gru_scan<<<1, 512, 0, stream>>>(gi, whh, bhh, ht);
  cond_kernel<<<64, 256, 0, stream>>>(ht, htp_w, htp_b, cond);
  qsample_kernel<<<4096, 256, 0, stream>>>(fut, noise, sab, s1ab, y0);

  // dilated temporal residual stack
  float* ycur = y0;
  float* ynxt = y1;
  for (int l = 0; l < NL; ++l) {
    GemmArgs gc{conv_w + (size_t)l * 2048 * 2048 * 3, conv_b + l * 2048,
                ycur, nullptr, hbuf, 2048, 6144, 1 << l, 0.f};
    wmma_gemm<MODE_CONV><<<dim3(16, 4), 256, 0, stream>>>(gc);
    GemmArgs gp{proj_w + (size_t)l * 2048 * 2048, proj_b + l * 2048,
                hbuf, ycur, ynxt, 2048, 2048, 0, 0.f};
    wmma_gemm<MODE_PROJ><<<dim3(16, 4), 256, 0, stream>>>(gp);
    float* t = ycur; ycur = ynxt; ynxt = t;
  }

  // GAT
  hei_kernel<<<64, 256, 0, stream>>>(ycur, gat_w, gat_a, Hm, ei, ej);
  maxej_kernel<<<8, 256, 0, stream>>>(ej, mxej);
  gat_kernel<<<dim3(16, 8), 256, 0, stream>>>(Hm, ei, ej, mxej, cond, Yb);

  // final 1x1 conv + fused MSE reduction
  {
    GemmArgs go{out_w, out_b, Yb, noise, loss, 2048, 2048, 0, 1.f / 1048576.f};
    wmma_gemm<MODE_OUT><<<dim3(16, 4), 256, 0, stream>>>(go);
  }
}
